// TreeHopModel_72610717106537
// MI455X (gfx1250) — matrix operand order repaired and checked by reading, not compiled, not run on hardware
//
#include <hip/hip_runtime.h>

// ---------------- problem constants ----------------
constexpr int N_NODES = 65536;
constexpr int E_EDGES = 16384;
constexpr int D_DIM   = 1024;
constexpr int A_DIM   = 512;
constexpr int M_DIM   = 2048;
constexpr int H_HEADS = 3;
constexpr int NRES    = 3;

// ---------------- WMMA / TDM types ----------------
typedef __attribute__((ext_vector_type(16))) __bf16 v16bf;
typedef __attribute__((ext_vector_type(8)))  float  v8f;
typedef __attribute__((ext_vector_type(4)))  unsigned int u32x4;
typedef __attribute__((ext_vector_type(8)))  unsigned int u32x8;

// ---------------- helpers ----------------
__device__ __forceinline__ unsigned short f2bf(float f) {
  unsigned int u = __float_as_uint(f);
  if ((u & 0x7fffffffu) > 0x7f800000u) return (unsigned short)0x7fc0u; // NaN
  unsigned int r = u + 0x7fffu + ((u >> 16) & 1u); // round-to-nearest-even
  return (unsigned short)(r >> 16);
}

__device__ __forceinline__ float warp_sum(float v) {
  #pragma unroll
  for (int o = 16; o > 0; o >>= 1) v += __shfl_xor(v, o, 32);
  return v;
}
__device__ __forceinline__ float warp_max(float v) {
  #pragma unroll
  for (int o = 16; o > 0; o >>= 1) v = fmaxf(v, __shfl_xor(v, o, 32));
  return v;
}

// 256-thread block sum (wave32: 8 waves)
__device__ __forceinline__ float block_sum(float v, float* sred) {
  v = warp_sum(v);
  __syncthreads();
  if ((threadIdx.x & 31) == 0) sred[threadIdx.x >> 5] = v;
  __syncthreads();
  float t = (threadIdx.x < 8) ? sred[threadIdx.x] : 0.f;
  t = warp_sum(t);
  if (threadIdx.x == 0) sred[0] = t;
  __syncthreads();
  return sred[0];
}

// ---------------- fp32 -> bf16 convert ----------------
__global__ __launch_bounds__(256) void f2bf_k(const float* __restrict__ in,
                                              unsigned short* __restrict__ out,
                                              size_t n) {
  size_t i = (size_t)blockIdx.x * blockDim.x + threadIdx.x;
  size_t stride = (size_t)gridDim.x * blockDim.x;
  for (; i < n; i += stride) out[i] = f2bf(in[i]);
}

// ---------------- gather + L2 normalize ----------------
__global__ __launch_bounds__(256) void gather_norm_k(
    const float* __restrict__ h, const float* __restrict__ rep,
    const int* __restrict__ src, const int* __restrict__ dst,
    float* __restrict__ Qf, float* __restrict__ Kf,
    unsigned short* __restrict__ Qb, unsigned short* __restrict__ Kb) {
  __shared__ float sred[8];
  const int e = blockIdx.x;
  const int t = threadIdx.x;
  const float* hr = h   + (size_t)src[e] * D_DIM;
  const float* kr = rep + (size_t)dst[e] * D_DIM;
  float v[4];
  float ss = 0.f;
  #pragma unroll
  for (int i = 0; i < 4; i++) { v[i] = hr[t + i * 256]; ss += v[i] * v[i]; }
  ss = block_sum(ss, sred);
  float sc = 1.f / fmaxf(sqrtf(ss), 1e-12f);
  #pragma unroll
  for (int i = 0; i < 4; i++) {
    int c = t + i * 256;
    size_t idx = (size_t)e * D_DIM + c;
    float q = v[i] * sc;
    float k = kr[c];
    Qf[idx] = q; Qb[idx] = f2bf(q);
    Kf[idx] = k; Kb[idx] = f2bf(k);
  }
}

// ---------------- softmax attention over feature dim A ----------------
__global__ __launch_bounds__(256) void attn_k(
    const float* __restrict__ Qp, const float* __restrict__ Kp,
    const float* __restrict__ Vp,
    float* __restrict__ aof, unsigned short* __restrict__ aob) {
  __shared__ float sred[8];
  const size_t row = blockIdx.x;
  const int t = threadIdx.x;
  const float scale = 0.04419417382415922f; // 1/sqrt(512)
  const float* q = Qp + row * A_DIM;
  const float* k = Kp + row * A_DIM;
  const float* vv = Vp + row * A_DIM;
  float s[2];
  float mx = -3.4e38f;
  #pragma unroll
  for (int i = 0; i < 2; i++) {
    int c = t + i * 256;
    s[i] = q[c] * k[c] * scale;
    mx = fmaxf(mx, s[i]);
  }
  mx = warp_max(mx);
  __syncthreads();
  if ((t & 31) == 0) sred[t >> 5] = mx;
  __syncthreads();
  float m2 = (t < 8) ? sred[t] : -3.4e38f;
  m2 = warp_max(m2);
  if (t == 0) sred[0] = m2;
  __syncthreads();
  mx = sred[0];
  float sum = 0.f;
  #pragma unroll
  for (int i = 0; i < 2; i++) { s[i] = expf(s[i] - mx); sum += s[i]; }
  sum = block_sum(sum, sred);
  float inv = 1.f / sum;
  #pragma unroll
  for (int i = 0; i < 2; i++) {
    int c = t + i * 256;
    float o = s[i] * inv * vv[c];
    aof[row * A_DIM + c] = o;
    aob[row * A_DIM + c] = f2bf(o);
  }
}

// ---------------- LayerNorm over M, writes bf16 xn ----------------
__global__ __launch_bounds__(256) void ln_k(
    const float* __restrict__ m, const float* __restrict__ g_ln,
    const float* __restrict__ b_ln, unsigned short* __restrict__ xn, int j) {
  __shared__ float sred[8];
  const size_t row = blockIdx.x;            // 0 .. H*E-1
  const int hh = (int)(row / E_EDGES);
  const int t = threadIdx.x;
  const float* x = m + row * M_DIM;
  const float* g = g_ln + ((size_t)hh * NRES + j) * M_DIM;
  const float* b = b_ln + ((size_t)hh * NRES + j) * M_DIM;
  float v[8];
  float s = 0.f;
  #pragma unroll
  for (int i = 0; i < 8; i++) { v[i] = x[t + i * 256]; s += v[i]; }
  float mu = block_sum(s, sred) * (1.f / M_DIM);
  float var = 0.f;
  #pragma unroll
  for (int i = 0; i < 8; i++) { float d = v[i] - mu; var += d * d; }
  var = block_sum(var, sred) * (1.f / M_DIM);
  float inv = rsqrtf(var + 1e-5f);
  #pragma unroll
  for (int i = 0; i < 8; i++) {
    int c = t + i * 256;
    xn[row * M_DIM + c] = f2bf((v[i] - mu) * inv * g[c] + b[c]);
  }
}

// ---------------- bf16 WMMA GEMM, double-buffered TDM pipeline ----------
// C[rows x N] = A[rows x K] * B[K x N]
// A tile (BMxBK row-major, stride lda) fetched by the Tensor Data Mover into
// LDS (2D D#: tile_dim0=BK, tile_dim1=BM, stride0=lda, data_size=2B).
// B tile staged by threads with packed-b32 transposed stores -> LDS [n][k].
// Double-buffered: TDM(t+1) + B(t+1) global loads fly under compute(t);
// per-wave in-order TENSORcnt lets wave0 use s_wait_tensorcnt 1.
// EPI 0: C = acc + bias                (fp32)
// EPI 1: C = aux0 + relu(acc + bias)   (fp32, residual)
// EPI 2: C = acc + bias + aux0         (bf16, strided cat)
// EPI 3: C = acc + aux0 - aux1         (fp32, final Q-K)
constexpr int BM = 128, BN = 128, BK = 32;

template <int EPI>
__global__ __launch_bounds__(256) void gemm_bf16_k(
    const unsigned short* __restrict__ Amat, int lda,
    const unsigned short* __restrict__ Bmat, int ldb,
    void* __restrict__ Cout, int ldc,
    const float* __restrict__ bias,
    const float* __restrict__ aux0, int ld0,
    const float* __restrict__ aux1, int ld1,
    int Kdim) {
  __shared__ unsigned short lA[2][BM * BK];  // [m][k]  (filled by TDM)
  __shared__ unsigned short lB[2][BN * BK];  // [n][k]  (transposed store)

  const int tid  = threadIdx.x;
  const int lane = tid & 31;
  const int warp = tid >> 5;
  const int rowBase = blockIdx.y * BM;
  const int colBase = blockIdx.x * BN;
  const int wRow = (warp >> 1) * 32;
  const int wCol = (warp & 1) * 64;
  const int kg = lane >> 4;   // 0/1 half-wave (K sub-range select)
  const int lm = lane & 15;

  v8f acc[2][4];
  #pragma unroll
  for (int i = 0; i < 2; i++)
    #pragma unroll
    for (int j = 0; j < 4; j++) acc[i][j] = (v8f){0.f,0.f,0.f,0.f,0.f,0.f,0.f,0.f};

  // B tile loader: thread handles k-pair kp (k=2*kp,2*kp+1), 8 columns
  const int bKp = tid >> 4;          // 0..15 -> k = 2*bKp
  const int bN  = (tid & 15) * 8;    // 0..120

  // TDM descriptor group 1 (tile-invariant); ISA 8.4 bit layout.
  const unsigned ldsA_off0 = (unsigned)(unsigned long long)(void*)&lA[0][0];
  const unsigned ldsA_off1 = (unsigned)(unsigned long long)(void*)&lA[1][0];
  const unsigned long long abase =
      (unsigned long long)(const void*)(Amat + (size_t)rowBase * lda);
  u32x8 g1;
  g1[0] = 1u << 16;                         // wg_mask=0, data_size=1 (2 bytes)
  g1[1] = 0u;                               // tensor_dim0 = 1<<30 (lo16=0)
  g1[2] = 0x4000u;                          // tensor_dim0 hi16, tensor_dim1 lo16=0
  g1[3] = ((unsigned)BK << 16) | 0x4000u;   // tensor_dim1 hi16, tile_dim0=BK
  g1[4] = (unsigned)BM;                     // tile_dim1=BM, tile_dim2=0
  g1[5] = (unsigned)lda;                    // tensor_dim0_stride lo32
  g1[6] = 0u;                               // stride0 hi16, stride1 lo16
  g1[7] = 0u;                               // stride1 hi32

  // issue TDM load of A tile t into lA[buf] (call from wave 0 only)
  auto tdmA = [&](int t, int buf) {
    unsigned long long ga = abase + (unsigned long long)t * (BK * 2ull);
    u32x4 g0;
    g0[0] = 1u;                                                // count=1
    g0[1] = buf ? ldsA_off1 : ldsA_off0;                       // lds_addr
    g0[2] = (unsigned)ga;                                      // global_addr lo
    g0[3] = ((unsigned)(ga >> 32) & 0x01FFFFFFu) | (2u << 30); // addr hi|type=2
    asm volatile("tensor_load_to_lds %0, %1" :: "s"(g0), "s"(g1) : "memory");
  };

  // stage B tile t into lB[buf] (transposed, packed b32 stores)
  auto stageB = [&](int t, int buf) {
    const unsigned short* gb =
        Bmat + (size_t)(t * BK + 2 * bKp) * ldb + colBase + bN;
    uint4 r0 = *reinterpret_cast<const uint4*>(gb);
    uint4 r1 = *reinterpret_cast<const uint4*>(gb + ldb);
    const unsigned short* u0 = reinterpret_cast<const unsigned short*>(&r0);
    const unsigned short* u1 = reinterpret_cast<const unsigned short*>(&r1);
    unsigned int* dst = reinterpret_cast<unsigned int*>(&lB[buf][0]);
    #pragma unroll
    for (int j = 0; j < 8; j++) {
      unsigned int pk = (unsigned int)u0[j] | ((unsigned int)u1[j] << 16);
      dst[((bN + j) * BK + 2 * bKp) >> 1] = pk;   // lB[n][k], k-pair packed
    }
  };

  // one 128x128x32 tile of WMMAs from LDS buffer `buf`
  auto computeTile = [&](int buf) {
    v16bf afr[2], bfr[4];
    #pragma unroll
    for (int i = 0; i < 2; i++) {
      const unsigned short* p = &lA[buf][(wRow + i * 16 + lm) * BK];
      union { v16bf v; uint4 q[2]; } u;
      u.q[0] = *reinterpret_cast<const uint4*>(p + kg * 8);       // K=kg*8..+7
      u.q[1] = *reinterpret_cast<const uint4*>(p + 16 + kg * 8);  // K=16+kg*8..
      afr[i] = u.v;
    }
    #pragma unroll
    for (int j = 0; j < 4; j++) {
      const unsigned short* p = &lB[buf][(wCol + j * 16 + lm) * BK];
      union { v16bf v; uint4 q[2]; } u;
      u.q[0] = *reinterpret_cast<const uint4*>(p + kg * 8);
      u.q[1] = *reinterpret_cast<const uint4*>(p + 16 + kg * 8);
      bfr[j] = u.v;
    }
    #pragma unroll
    for (int i = 0; i < 2; i++)
      #pragma unroll
      for (int j = 0; j < 4; j++)
        acc[i][j] = __builtin_amdgcn_wmma_f32_16x16x32_bf16(
            false, afr[i], false, bfr[j], (short)0, acc[i][j], false, false);
  };

  const int T = Kdim / BK;
  if (warp == 0) tdmA(0, 0);
  stageB(0, 0);
  for (int t = 0; t < T; t++) {
    const int cur = t & 1;
    const int nxt = cur ^ 1;
    if (t + 1 < T) {                 // prefetch next tile under compute
      if (warp == 0) tdmA(t + 1, nxt);
      stageB(t + 1, nxt);
    }
    if (warp == 0) {
      if (t + 1 < T) __builtin_amdgcn_s_wait_tensorcnt(1);  // TDM(t) done
      else           __builtin_amdgcn_s_wait_tensorcnt(0);
    }
    __syncthreads();                 // lA[cur]/lB[cur] ready for all waves
    computeTile(cur);
    __syncthreads();                 // all reads of cur done before overwrite
  }

  // epilogue: element r of v8f -> m = kg*8 + r, n = lm
  #pragma unroll
  for (int i = 0; i < 2; i++) {
    #pragma unroll
    for (int j = 0; j < 4; j++) {
      #pragma unroll
      for (int r = 0; r < 8; r++) {
        int grow = rowBase + wRow + i * 16 + kg * 8 + r;
        int gcol = colBase + wCol + j * 16 + lm;
        float v = acc[i][j][r];
        if (EPI == 0) {
          v += bias[gcol];
          reinterpret_cast<float*>(Cout)[(size_t)grow * ldc + gcol] = v;
        } else if (EPI == 1) {
          v += bias[gcol];
          v = v > 0.f ? v : 0.f;
          v += aux0[(size_t)grow * ld0 + gcol];
          reinterpret_cast<float*>(Cout)[(size_t)grow * ldc + gcol] = v;
        } else if (EPI == 2) {
          v += bias[gcol] + aux0[(size_t)grow * ld0 + gcol];
          reinterpret_cast<unsigned short*>(Cout)[(size_t)grow * ldc + gcol] = f2bf(v);
        } else {
          v += aux0[(size_t)grow * ld0 + gcol] - aux1[(size_t)grow * ld1 + gcol];
          reinterpret_cast<float*>(Cout)[(size_t)grow * ldc + gcol] = v;
        }
      }
    }
  }
}

// ---------------- host orchestration ----------------
extern "C" void kernel_launch(void* const* d_in, const int* in_sizes, int n_in,
                              void* d_out, int out_size, void* d_ws, size_t ws_size,
                              hipStream_t stream) {
  const float* h    = (const float*)d_in[0];
  const float* rep  = (const float*)d_in[1];
  const int*   src  = (const int*)d_in[2];
  const int*   dst  = (const int*)d_in[3];
  const float* Wq   = (const float*)d_in[4];
  const float* bq   = (const float*)d_in[5];
  const float* Wk   = (const float*)d_in[6];
  const float* bk   = (const float*)d_in[7];
  const float* Wv   = (const float*)d_in[8];
  const float* bv   = (const float*)d_in[9];
  const float* W0   = (const float*)d_in[10];
  const float* b0   = (const float*)d_in[11];
  const float* Wr   = (const float*)d_in[12];
  const float* br   = (const float*)d_in[13];
  const float* g_ln = (const float*)d_in[14];
  const float* b_ln = (const float*)d_in[15];
  const float* Wms  = (const float*)d_in[16];
  const float* bms  = (const float*)d_in[17];
  const float* Wuas = (const float*)d_in[18];
  (void)in_sizes; (void)n_in; (void)out_size; (void)ws_size;

  char* ws = (char*)d_ws;
  size_t off = 0;
  auto alloc = [&](size_t bytes) -> char* {
    char* p = ws + off;
    off = (off + bytes + 255) & ~(size_t)255;
    return p;
  };

  const size_t HDA = (size_t)H_HEADS * D_DIM * A_DIM;
  const size_t HAM = (size_t)H_HEADS * A_DIM * M_DIM;
  const size_t HRMM = (size_t)H_HEADS * NRES * M_DIM * M_DIM;
  const size_t HMA = (size_t)H_HEADS * M_DIM * A_DIM;
  const size_t HAD = (size_t)H_HEADS * A_DIM * D_DIM;
  const size_t ED  = (size_t)E_EDGES * D_DIM;
  const size_t HEA = (size_t)H_HEADS * E_EDGES * A_DIM;
  const size_t HEM = (size_t)H_HEADS * E_EDGES * M_DIM;

  unsigned short* Wq_b   = (unsigned short*)alloc(HDA * 2);
  unsigned short* Wk_b   = (unsigned short*)alloc(HDA * 2);
  unsigned short* Wv_b   = (unsigned short*)alloc(HDA * 2);
  unsigned short* W0_b   = (unsigned short*)alloc(HAM * 2);
  unsigned short* Wr_b   = (unsigned short*)alloc(HRMM * 2);
  unsigned short* Wms_b  = (unsigned short*)alloc(HMA * 2);
  unsigned short* Wuas_b = (unsigned short*)alloc(HAD * 2);
  float*          Qf     = (float*)alloc(ED * 4);
  float*          Kf     = (float*)alloc(ED * 4);
  unsigned short* Qb     = (unsigned short*)alloc(ED * 2);
  unsigned short* Kb     = (unsigned short*)alloc(ED * 2);
  float*          Qp     = (float*)alloc(HEA * 4);
  float*          Kp     = (float*)alloc(HEA * 4);
  float*          Vp     = (float*)alloc(HEA * 4);
  float*          aof    = (float*)alloc(HEA * 4);
  unsigned short* aob    = (unsigned short*)alloc(HEA * 2);
  float*          mbuf   = (float*)alloc(HEM * 4);
  unsigned short* xn     = (unsigned short*)alloc(HEM * 2);
  unsigned short* mb     = (unsigned short*)alloc(HEM * 2);
  unsigned short* cat    = (unsigned short*)alloc((size_t)E_EDGES * H_HEADS * A_DIM * 2);

  dim3 blk(256);

  // 1) weights -> bf16
  f2bf_k<<<2048, blk, 0, stream>>>(Wq,   Wq_b,   HDA);
  f2bf_k<<<2048, blk, 0, stream>>>(Wk,   Wk_b,   HDA);
  f2bf_k<<<2048, blk, 0, stream>>>(Wv,   Wv_b,   HDA);
  f2bf_k<<<2048, blk, 0, stream>>>(W0,   W0_b,   HAM);
  f2bf_k<<<8192, blk, 0, stream>>>(Wr,   Wr_b,   HRMM);
  f2bf_k<<<2048, blk, 0, stream>>>(Wms,  Wms_b,  HMA);
  f2bf_k<<<2048, blk, 0, stream>>>(Wuas, Wuas_b, HAD);

  // 2) gather + normalize
  gather_norm_k<<<E_EDGES, blk, 0, stream>>>(h, rep, src, dst, Qf, Kf, Qb, Kb);

  // 3) per-head Q/K/V projections: [E,1024] x [1024,512]
  for (int hh = 0; hh < H_HEADS; hh++) {
    dim3 g(A_DIM / BN, E_EDGES / BM);
    gemm_bf16_k<0><<<g, blk, 0, stream>>>(
        Qb, D_DIM, Wq_b + (size_t)hh * D_DIM * A_DIM, A_DIM,
        Qp + (size_t)hh * E_EDGES * A_DIM, A_DIM,
        bq + (size_t)hh * A_DIM, nullptr, 0, nullptr, 0, D_DIM);
    gemm_bf16_k<0><<<g, blk, 0, stream>>>(
        Kb, D_DIM, Wk_b + (size_t)hh * D_DIM * A_DIM, A_DIM,
        Kp + (size_t)hh * E_EDGES * A_DIM, A_DIM,
        bk + (size_t)hh * A_DIM, nullptr, 0, nullptr, 0, D_DIM);
    gemm_bf16_k<0><<<g, blk, 0, stream>>>(
        Kb, D_DIM, Wv_b + (size_t)hh * D_DIM * A_DIM, A_DIM,
        Vp + (size_t)hh * E_EDGES * A_DIM, A_DIM,
        bv + (size_t)hh * A_DIM, nullptr, 0, nullptr, 0, D_DIM);
  }

  // 4) feature-dim softmax attention
  attn_k<<<H_HEADS * E_EDGES, blk, 0, stream>>>(Qp, Kp, Vp, aof, aob);

  // 5) m = attn_out @ W0 + b0 : [E,512] x [512,2048]
  for (int hh = 0; hh < H_HEADS; hh++) {
    dim3 g(M_DIM / BN, E_EDGES / BM);
    gemm_bf16_k<0><<<g, blk, 0, stream>>>(
        aob + (size_t)hh * E_EDGES * A_DIM, A_DIM,
        W0_b + (size_t)hh * A_DIM * M_DIM, M_DIM,
        mbuf + (size_t)hh * E_EDGES * M_DIM, M_DIM,
        b0 + (size_t)hh * M_DIM, nullptr, 0, nullptr, 0, A_DIM);
  }

  // 6) NRES residual blocks: m += relu(LN(m) @ Wr + br)
  for (int j = 0; j < NRES; j++) {
    ln_k<<<H_HEADS * E_EDGES, blk, 0, stream>>>(mbuf, g_ln, b_ln, xn, j);
    for (int hh = 0; hh < H_HEADS; hh++) {
      dim3 g(M_DIM / BN, E_EDGES / BM);
      gemm_bf16_k<1><<<g, blk, 0, stream>>>(
          xn + (size_t)hh * E_EDGES * M_DIM, M_DIM,
          Wr_b + ((size_t)hh * NRES + j) * M_DIM * M_DIM, M_DIM,
          mbuf + (size_t)hh * E_EDGES * M_DIM, M_DIM,
          br + ((size_t)hh * NRES + j) * M_DIM,
          mbuf + (size_t)hh * E_EDGES * M_DIM, M_DIM, nullptr, 0, M_DIM);
    }
  }

  // 7) m -> bf16
  f2bf_k<<<8192, blk, 0, stream>>>(mbuf, mb, HEM);

  // 8) head_out = m @ Wms + bms + attn_out, written bf16 into cat[E, H*A]
  for (int hh = 0; hh < H_HEADS; hh++) {
    dim3 g(A_DIM / BN, E_EDGES / BM);
    gemm_bf16_k<2><<<g, blk, 0, stream>>>(
        mb + (size_t)hh * E_EDGES * M_DIM, M_DIM,
        Wms_b + (size_t)hh * M_DIM * A_DIM, A_DIM,
        cat + (size_t)hh * A_DIM, H_HEADS * A_DIM,
        bms + (size_t)hh * A_DIM,
        aof + (size_t)hh * E_EDGES * A_DIM, A_DIM, nullptr, 0, M_DIM);
  }

  // 9) out = Q - K + cat @ Wuas : [E,1536] x [1536,1024]
  {
    dim3 g(D_DIM / BN, E_EDGES / BM);
    gemm_bf16_k<3><<<g, blk, 0, stream>>>(
        cat, H_HEADS * A_DIM, Wuas_b, D_DIM,
        d_out, D_DIM, nullptr, Qf, D_DIM, Kf, D_DIM, H_HEADS * A_DIM);
  }
}
